// MessagePassingLayer_43817256354334
// MI455X (gfx1250) — compile-verified, matrix operand order checked by reading, not run on hardware
//
#include <hip/hip_runtime.h>
#include <hip/hip_bf16.h>

typedef __attribute__((ext_vector_type(16))) __bf16 v16bf;
typedef __attribute__((ext_vector_type(8)))  __bf16 v8bf;
typedef __attribute__((ext_vector_type(8)))  float  v8f;

#define HID 64

// convert 8 consecutive f32 -> 8 bf16 (two aligned float4 loads)
__device__ __forceinline__ v8bf cvt8(const float* p) {
    float4 a = *(const float4*)p;
    float4 b = *(const float4*)(p + 4);
    v8bf r;
    r[0] = (__bf16)a.x; r[1] = (__bf16)a.y; r[2] = (__bf16)a.z; r[3] = (__bf16)a.w;
    r[4] = (__bf16)b.x; r[5] = (__bf16)b.y; r[6] = (__bf16)b.z; r[7] = (__bf16)b.w;
    return r;
}

__device__ __forceinline__ v16bf pack16(v8bf lo, v8bf hi) {
    v16bf r;
    #pragma unroll
    for (int i = 0; i < 8; ++i) { r[i] = lo[i]; r[i + 8] = hi[i]; }
    return r;
}

// 16-bit matrix fragment (A or B role; both use the same lane->element map):
// lane: row/col index = base + ln ; kbase = hi*8 ; elems 0..7 -> k=c*32+kb+0..7,
// elems 8..15 -> k=c*32+kb+16..23.  Source is LDS-staged W^T [n][k], stride KP.
__device__ __forceinline__ v16bf load_wfrag(const __bf16* wt, int KP, int n, int c, int kb) {
    const __bf16* p = wt + n * KP + c * 32 + kb;
    v8bf lo = *(const v8bf*)p;
    v8bf hi = *(const v8bf*)(p + 16);
    return pack16(lo, hi);
}

__device__ __forceinline__ float silu(float v) {
    return v / (1.0f + __expf(-v));
}

// guaranteed single-instruction fp32 scatter-add (no CAS loop)
__device__ __forceinline__ void atomic_add_f32(float* p, float v) {
    asm volatile("global_atomic_add_f32 %0, %1, off" :: "v"(p), "v"(v) : "memory");
}

// ---------------- edge MLP + scatter-add ----------------
__global__ void __launch_bounds__(128)
edge_mlp_kernel(const float* __restrict__ x, const int* __restrict__ ei,
                const float* __restrict__ ea,
                const float* __restrict__ W1, const float* __restrict__ b1,
                const float* __restrict__ W2, const float* __restrict__ b2,
                float* __restrict__ agg, int E, int nTiles)
{
    __shared__ __bf16 w1t[64 * 160];   // W1^T, K padded 144->160, [n][k]
    __shared__ __bf16 w2t[64 * 64];    // W2^T [n][k]

    const int tid = threadIdx.x;
    for (int i = tid; i < 64 * 160; i += blockDim.x) {
        int n = i / 160, k = i % 160;
        w1t[i] = (k < 144) ? (__bf16)W1[k * 64 + n] : (__bf16)0.0f;
    }
    for (int i = tid; i < 64 * 64; i += blockDim.x) {
        int n = i >> 6, k = i & 63;
        w2t[i] = (__bf16)W2[k * 64 + n];
    }
    __syncthreads();

    const int lane = tid & 31;
    const int wid  = tid >> 5;
    const int ln   = lane & 15;      // edge-row (B/D col) or fragment row
    const int hi   = lane >> 4;      // lane-group
    const int kb   = hi * 8;         // within-chunk k base / D-row base
    const int nwv  = gridDim.x * (blockDim.x >> 5);
    const int gw   = blockIdx.x * (blockDim.x >> 5) + wid;

    // hoisted biases (tile-invariant)
    float4 b1lo[4], b1hi[4];
    #pragma unroll
    for (int mt = 0; mt < 4; ++mt) {
        b1lo[mt] = *(const float4*)(b1 + mt * 16 + kb);
        b1hi[mt] = *(const float4*)(b1 + mt * 16 + kb + 4);
    }
    float b2v[4];
    #pragma unroll
    for (int nt = 0; nt < 4; ++nt) b2v[nt] = b2[nt * 16 + ln];

    for (int t = gw; t < nTiles; t += nwv) {
        const int e = t * 16 + ln;
        const int s = ei[e];
        const int d = ei[E + e];
        const float* xs = x + (long)s * HID;
        const float* xd = x + (long)d * HID;

        // B fragments of m_in^T: 160x16 bf16 (5 chunks of K=32)
        v16bf Bv[5];
        Bv[0] = pack16(cvt8(xs + kb),      cvt8(xs + kb + 16));
        Bv[1] = pack16(cvt8(xs + 32 + kb), cvt8(xs + 32 + kb + 16));
        Bv[2] = pack16(cvt8(xd + kb),      cvt8(xd + kb + 16));
        Bv[3] = pack16(cvt8(xd + 32 + kb), cvt8(xd + 32 + kb + 16));
        {
            v8bf lo = cvt8(ea + (long)e * 16 + kb); // cols 0..7 / 8..15
            v8bf z;
            #pragma unroll
            for (int i = 0; i < 8; ++i) z[i] = (__bf16)0.0f; // pad K 144..159
            Bv[4] = pack16(lo, z);
        }

        // GEMM1 (transposed): h^T = W1^T @ m_in^T.  D-frag of h^T per lane is
        // exactly the A-frag of h that GEMM2 needs -> pack directly, no LDS.
        v16bf A2[2];
        #pragma unroll
        for (int mt = 0; mt < 4; ++mt) {
            v8f acc;
            acc[0] = b1lo[mt].x; acc[1] = b1lo[mt].y; acc[2] = b1lo[mt].z; acc[3] = b1lo[mt].w;
            acc[4] = b1hi[mt].x; acc[5] = b1hi[mt].y; acc[6] = b1hi[mt].z; acc[7] = b1hi[mt].w;
            #pragma unroll
            for (int c = 0; c < 5; ++c) {
                v16bf Wf = load_wfrag(w1t, 160, mt * 16 + ln, c, kb);
                acc = __builtin_amdgcn_wmma_f32_16x16x32_bf16(
                          false, Wf, false, Bv[c], (short)0, acc, false, false);
            }
            #pragma unroll
            for (int r = 0; r < 8; ++r)
                A2[mt >> 1][(mt & 1) * 8 + r] = (__bf16)silu(acc[r]);
        }

        // destination rows for this lane-group's D-fragment rows (m = kb + r)
        int4 d0 = *(const int4*)(ei + E + t * 16 + kb);
        int4 d1 = *(const int4*)(ei + E + t * 16 + kb + 4);
        int drow[8] = {d0.x, d0.y, d0.z, d0.w, d1.x, d1.y, d1.z, d1.w};

        // GEMM2 (normal): msg = h @ W2 ; coalesced scatter-add into agg
        #pragma unroll
        for (int nt = 0; nt < 4; ++nt) {
            v8f acc;
            #pragma unroll
            for (int j = 0; j < 8; ++j) acc[j] = b2v[nt];
            #pragma unroll
            for (int c = 0; c < 2; ++c) {
                v16bf Wf = load_wfrag(w2t, 64, nt * 16 + ln, c, kb);
                acc = __builtin_amdgcn_wmma_f32_16x16x32_bf16(
                          false, A2[c], false, Wf, (short)0, acc, false, false);
            }
            #pragma unroll
            for (int r = 0; r < 8; ++r) {
                float v = silu(acc[r]);
                atomic_add_f32(&agg[(long)drow[r] * HID + nt * 16 + ln], v);
            }
        }
    }
}

// ---------------- node MLP (computed transposed for vector stores) ----------------
__global__ void __launch_bounds__(128)
node_mlp_kernel(const float* __restrict__ x, const float* __restrict__ agg,
                const float* __restrict__ W3, const float* __restrict__ b3,
                float* __restrict__ out, int nTiles)
{
    __shared__ __bf16 w3t[64 * 128]; // W3^T [n][k], K=128

    const int tid = threadIdx.x;
    for (int i = tid; i < 64 * 128; i += blockDim.x) {
        int n = i >> 7, k = i & 127;
        w3t[i] = (__bf16)W3[k * 64 + n];
    }
    __syncthreads();

    const int lane = tid & 31;
    const int wid  = tid >> 5;
    const int ln   = lane & 15;
    const int hi   = lane >> 4;
    const int kb   = hi * 8;
    const int nwv  = gridDim.x * (blockDim.x >> 5);
    const int gw   = blockIdx.x * (blockDim.x >> 5) + wid;

    float4 b3lo[4], b3hi[4];
    #pragma unroll
    for (int mt = 0; mt < 4; ++mt) {
        b3lo[mt] = *(const float4*)(b3 + mt * 16 + kb);
        b3hi[mt] = *(const float4*)(b3 + mt * 16 + kb + 4);
    }

    for (int t = gw; t < nTiles; t += nwv) {
        const int row = t * 16 + ln;
        const float* xr = x   + (long)row * HID;
        const float* ar = agg + (long)row * HID;

        // B fragments of node_in^T (128x16)
        v16bf Bv[4];
        Bv[0] = pack16(cvt8(xr + kb),      cvt8(xr + kb + 16));
        Bv[1] = pack16(cvt8(xr + 32 + kb), cvt8(xr + 32 + kb + 16));
        Bv[2] = pack16(cvt8(ar + kb),      cvt8(ar + kb + 16));
        Bv[3] = pack16(cvt8(ar + 32 + kb), cvt8(ar + 32 + kb + 16));

        // out^T = W3^T @ node_in^T : each lane ends up with 8 consecutive
        // output features of its own node -> two b128 stores per m-tile.
        float* orow = out + (long)row * HID;
        #pragma unroll
        for (int mt = 0; mt < 4; ++mt) {
            v8f acc;
            acc[0] = b3lo[mt].x; acc[1] = b3lo[mt].y; acc[2] = b3lo[mt].z; acc[3] = b3lo[mt].w;
            acc[4] = b3hi[mt].x; acc[5] = b3hi[mt].y; acc[6] = b3hi[mt].z; acc[7] = b3hi[mt].w;
            #pragma unroll
            for (int c = 0; c < 4; ++c) {
                v16bf Wf = load_wfrag(w3t, 128, mt * 16 + ln, c, kb);
                acc = __builtin_amdgcn_wmma_f32_16x16x32_bf16(
                          false, Wf, false, Bv[c], (short)0, acc, false, false);
            }
            float4 o0, o1;
            o0.x = silu(acc[0]); o0.y = silu(acc[1]); o0.z = silu(acc[2]); o0.w = silu(acc[3]);
            o1.x = silu(acc[4]); o1.y = silu(acc[5]); o1.z = silu(acc[6]); o1.w = silu(acc[7]);
            *(float4*)(orow + mt * 16 + kb)     = o0;
            *(float4*)(orow + mt * 16 + kb + 4) = o1;
        }
    }
}

extern "C" void kernel_launch(void* const* d_in, const int* in_sizes, int n_in,
                              void* d_out, int out_size, void* d_ws, size_t ws_size,
                              hipStream_t stream) {
    const float* x  = (const float*)d_in[0];
    const int*   ei = (const int*)  d_in[1];
    const float* ea = (const float*)d_in[2];
    const float* W1 = (const float*)d_in[3];
    const float* b1 = (const float*)d_in[4];
    const float* W2 = (const float*)d_in[5];
    const float* b2 = (const float*)d_in[6];
    const float* W3 = (const float*)d_in[7];
    const float* b3 = (const float*)d_in[8];
    float* out = (float*)d_out;

    const int N = in_sizes[0] / HID;      // 100000
    const int E = in_sizes[1] / 2;        // 1600000
    const int tilesE = E / 16;
    const int tilesN = N / 16;

    float* agg = (float*)d_ws;            // N*HID f32 scratch
    hipMemsetAsync(agg, 0, (size_t)N * HID * sizeof(float), stream);

    int blocksE = 1024;
    int maxBE = (tilesE + 3) / 4;
    if (blocksE > maxBE) blocksE = maxBE;
    edge_mlp_kernel<<<blocksE, 128, 0, stream>>>(x, ei, ea, W1, b1, W2, b2,
                                                 agg, E, tilesE);

    int blocksN = 256;
    int maxBN = (tilesN + 3) / 4;
    if (blocksN > maxBN) blocksN = maxBN;
    node_mlp_kernel<<<blocksN, 128, 0, stream>>>(x, agg, W3, b3, out, tilesN);
}
